// RnnGnnAgent_2800318677702
// MI455X (gfx1250) — matrix operand order repaired and checked by reading, not compiled, not run on hardware
//
#include <hip/hip_runtime.h>
#include <hip/hip_bf16.h>

// ---------------- problem constants ----------------
#define NND     100000
#define NEDGE   1600000
#define IN_DIM  128
#define HIDV    64
#define H2      128          // 2*HID (GRU hidden)
#define G3      384          // 3*H2
#define NACT    14
#define EDIM    5
#define NEG_SLOPE 0.2f

// bf16 weight region offsets (elements)
#define OFF_W1   0           // 64*128   = 8192
#define OFF_WIH  8192        // 384*64   = 24576
#define OFF_WHH  32768       // 384*128  = 49152
#define OFF_WL   81920       // 64*128   = 8192
#define OFF_WR   90112       // 64*128   = 8192
#define WBF_TOT  98304       // *2 bytes = 196608 (256B aligned)

typedef __attribute__((ext_vector_type(16))) __bf16 v16bf;
typedef __attribute__((ext_vector_type(8)))  float  v8f;

// ---------------- WMMA helpers (layouts per CDNA5 ISA §7.12.2) ----
static __device__ __forceinline__ v8f wmma_bf16(v16bf a, v16bf b, v8f c) {
    return __builtin_amdgcn_wmma_f32_16x16x32_bf16(
        /*neg_a=*/false, a, /*neg_b=*/false, b,
        /*c_mod=*/(short)0, c, /*reuse_a=*/false, /*reuse_b=*/false);
}

// A-frag (16x32 bf16) from row-major f32, rows start at `base`, leading dim ld,
// k-chunk already folded into base. Lane: row = lane&15; K set = {8g..8g+7} U {8g+16..8g+23}.
static __device__ __forceinline__ v16bf load_a_f32(const float* __restrict__ base,
                                                   int ld, int lane) {
    int m = lane & 15, grp = lane >> 4;
    const float* p = base + (size_t)m * ld + 8 * grp;
    v16bf a;
#pragma unroll
    for (int i = 0; i < 8; ++i) a[i]     = (__bf16)p[i];
#pragma unroll
    for (int i = 0; i < 8; ++i) a[8 + i] = (__bf16)p[16 + i];
    return a;
}

// A-frag from row-major bf16 in LDS.
static __device__ __forceinline__ v16bf load_a_lds(const __bf16* base, int ld, int lane) {
    int m = lane & 15, grp = lane >> 4;
    const __bf16* p = base + m * ld + 8 * grp;
    v16bf a;
#pragma unroll
    for (int i = 0; i < 8; ++i) a[i]     = p[i];
#pragma unroll
    for (int i = 0; i < 8; ++i) a[8 + i] = p[16 + i];
    return a;
}

// B-frag (32x16 bf16) for D = A * W^T : B[k][n] = W[n][k], W row-major [Nout, K] bf16.
// Lane: col n = nb + (lane&15); K set = kc + {8g..8g+7} U {8g+16..8g+23}.
static __device__ __forceinline__ v16bf load_b(const __bf16* __restrict__ W,
                                               int K, int nb, int kc, int lane) {
    int n = nb + (lane & 15), grp = lane >> 4;
    const __bf16* p = W + (size_t)n * K + kc + 8 * grp;
    v16bf b;
#pragma unroll
    for (int i = 0; i < 8; ++i) b[i] = p[i];
#pragma unroll
    for (int i = 0; i < 8; ++i) b[8 + i] = p[16 + i];
    return b;
}

static __device__ __forceinline__ float sigmoidf_(float x) {
    return 1.0f / (1.0f + __expf(-x));
}

// orderable-uint mapping for float atomic max
static __device__ __forceinline__ unsigned fkey(float f) {
    unsigned u = __float_as_uint(f);
    return (u & 0x80000000u) ? ~u : (u | 0x80000000u);
}
static __device__ __forceinline__ float funkey(unsigned k) {
    unsigned u = (k & 0x80000000u) ? (k & 0x7FFFFFFFu) : ~k;
    return __uint_as_float(u);
}

// ---------------- kernel 1: f32 -> bf16 weight repack -------------------------
__global__ void prep_weights(const float* __restrict__ W1, const float* __restrict__ w_ih,
                             const float* __restrict__ w_hh, const float* __restrict__ Wl,
                             const float* __restrict__ Wr, __bf16* __restrict__ wbf) {
    int i = blockIdx.x * blockDim.x + threadIdx.x;
    if (i >= WBF_TOT) return;
    float v;
    if      (i < OFF_WIH) v = W1[i - OFF_W1];
    else if (i < OFF_WHH) v = w_ih[i - OFF_WIH];
    else if (i < OFF_WL)  v = w_hh[i - OFF_WHH];
    else if (i < OFF_WR)  v = Wl[i - OFF_WL];
    else                  v = Wr[i - OFF_WR];
    wbf[i] = (__bf16)v;
}

// ---------------- kernel 2: init scratch accumulators -------------------------
__global__ void init_ws(float* __restrict__ agg, float* __restrict__ deg,
                        float* __restrict__ loop_attr, float* __restrict__ denom,
                        unsigned* __restrict__ mxkey, int n) {
    int i = blockIdx.x * blockDim.x + threadIdx.x;
    if (i < n * HIDV) agg[i] = 0.0f;
    if (i < n * EDIM) loop_attr[i] = 0.0f;
    if (i < n) { deg[i] = 0.0f; denom[i] = 0.0f; mxkey[i] = 0u; }  // 0u < fkey(-inf+eps)
}

// ---------------- kernel 3: fused node GEMMs (fc1 + GRU + xl/xr) via WMMA -----
__global__ __launch_bounds__(256) void node_kernel(
    const float* __restrict__ inputs, const float* __restrict__ hidden,
    const float* __restrict__ b1, const float* __restrict__ b_ih,
    const float* __restrict__ b_hh, const float* __restrict__ bl,
    const float* __restrict__ br, const __bf16* __restrict__ wbf,
    float* __restrict__ hs_out, float* __restrict__ xl, float* __restrict__ xr,
    int nTiles)
{
    const __bf16* W1b = wbf + OFF_W1;
    const __bf16* Wih = wbf + OFF_WIH;
    const __bf16* Whh = wbf + OFF_WHH;
    const __bf16* Wlb = wbf + OFF_WL;
    const __bf16* Wrb = wbf + OFF_WR;

    // per-wave LDS: x (16x64 bf16) + h_s (16x128 bf16) = 3072 elems = 6KB; 8 waves = 48KB
    __shared__ __bf16 lds[8 * (16 * HIDV + 16 * H2)];
    const int wave = threadIdx.x >> 5;
    const int lane = threadIdx.x & 31;
    const int tile = blockIdx.x * 8 + wave;
    const bool active = tile < nTiles;
    __bf16* xls = lds + wave * (16 * HIDV + 16 * H2);
    __bf16* hss = xls + 16 * HIDV;
    const int row0 = tile * 16;
    const int m = lane & 15, grp = lane >> 4;

    v16bf ax[2], ah[4];

    if (active) {
        // ---- stage 1: x = relu(inputs @ W1^T + b1)  [16 x 64] ----
        v16bf ain[4];
#pragma unroll
        for (int kf = 0; kf < 4; ++kf)
            ain[kf] = load_a_f32(inputs + (size_t)row0 * IN_DIM + 32 * kf, IN_DIM, lane);
#pragma unroll
        for (int nt = 0; nt < 4; ++nt) {
            v8f acc = {};
#pragma unroll
            for (int kf = 0; kf < 4; ++kf)
                acc = wmma_bf16(ain[kf], load_b(W1b, IN_DIM, nt * 16, 32 * kf, lane), acc);
            float bias = b1[nt * 16 + m];
#pragma unroll
            for (int r = 0; r < 8; ++r) {
                float v = acc[r] + bias;
                v = v > 0.0f ? v : 0.0f;
                xls[(8 * grp + r) * HIDV + nt * 16 + m] = (__bf16)v;   // row-major for A-frags
            }
        }
    }
    __syncthreads();
    if (active) {
        // ---- stage 2: GRU cell -> h_s [16 x 128] ----
#pragma unroll
        for (int kf = 0; kf < 2; ++kf) ax[kf] = load_a_lds(xls + 32 * kf, HIDV, lane);
#pragma unroll
        for (int kf = 0; kf < 4; ++kf)
            ah[kf] = load_a_f32(hidden + (size_t)row0 * H2 + 32 * kf, H2, lane);

        for (int t = 0; t < 8; ++t) {      // 8 column tiles of 16 over H2=128
            const int j0 = 16 * t;
            v8f air = {}, aiz = {}, ain_ = {}, ahr = {}, ahz = {}, ahn = {};
#pragma unroll
            for (int kf = 0; kf < 2; ++kf) {
                air  = wmma_bf16(ax[kf], load_b(Wih, HIDV, j0,        32 * kf, lane), air);
                aiz  = wmma_bf16(ax[kf], load_b(Wih, HIDV, H2 + j0,   32 * kf, lane), aiz);
                ain_ = wmma_bf16(ax[kf], load_b(Wih, HIDV, 2*H2 + j0, 32 * kf, lane), ain_);
            }
#pragma unroll
            for (int kf = 0; kf < 4; ++kf) {
                ahr = wmma_bf16(ah[kf], load_b(Whh, H2, j0,        32 * kf, lane), ahr);
                ahz = wmma_bf16(ah[kf], load_b(Whh, H2, H2 + j0,   32 * kf, lane), ahz);
                ahn = wmma_bf16(ah[kf], load_b(Whh, H2, 2*H2 + j0, 32 * kf, lane), ahn);
            }
            const int j = j0 + m;
            const float bir = b_ih[j], biz = b_ih[H2 + j], bin = b_ih[2 * H2 + j];
            const float bhr = b_hh[j], bhz = b_hh[H2 + j], bhn = b_hh[2 * H2 + j];
#pragma unroll
            for (int r = 0; r < 8; ++r) {
                const int row = row0 + 8 * grp + r;
                float hprev = hidden[(size_t)row * H2 + j];
                float rg = sigmoidf_(air[r] + bir + ahr[r] + bhr);
                float zg = sigmoidf_(aiz[r] + biz + ahz[r] + bhz);
                float ng = tanhf(ain_[r] + bin + rg * (ahn[r] + bhn));
                float hs = (1.0f - zg) * ng + zg * hprev;
                hs_out[(size_t)row * H2 + j] = hs;                 // output #2 (h_s)
                hss[(8 * grp + r) * H2 + j] = (__bf16)hs;
            }
        }
    }
    __syncthreads();
    if (active) {
        // ---- stage 3: xl = h_s @ Wl^T + bl, xr = h_s @ Wr^T + br ----
        v16bf ahs[4];
#pragma unroll
        for (int kf = 0; kf < 4; ++kf) ahs[kf] = load_a_lds(hss + 32 * kf, H2, lane);
#pragma unroll
        for (int nt = 0; nt < 4; ++nt) {
            v8f al = {}, ar = {};
#pragma unroll
            for (int kf = 0; kf < 4; ++kf) {
                al = wmma_bf16(ahs[kf], load_b(Wlb, H2, nt * 16, 32 * kf, lane), al);
                ar = wmma_bf16(ahs[kf], load_b(Wrb, H2, nt * 16, 32 * kf, lane), ar);
            }
            const float bL = bl[nt * 16 + m], bR = br[nt * 16 + m];
#pragma unroll
            for (int r = 0; r < 8; ++r) {
                const int row = row0 + 8 * grp + r;
                xl[(size_t)row * HIDV + nt * 16 + m] = al[r] + bL;
                xr[(size_t)row * HIDV + nt * 16 + m] = ar[r] + bR;
            }
        }
    }
}

// ---------------- kernel 4: degree + incoming edge-attr sums ------------------
__global__ void edge_deg_kernel(const int* __restrict__ dst, const float* __restrict__ eattr,
                                float* __restrict__ deg, float* __restrict__ loop_attr, int E) {
    int i = blockIdx.x * blockDim.x + threadIdx.x;
    if (i >= E) return;
    int d = dst[i];
    atomicAdd(deg + d, 1.0f);
#pragma unroll
    for (int c = 0; c < EDIM; ++c)
        atomicAdd(loop_attr + (size_t)d * EDIM + c, eattr[(size_t)i * EDIM + c]);
}

// ---------------- kernel 5: per-edge logits + segment max ---------------------
__global__ void edge_logits_kernel(const int* __restrict__ src, const int* __restrict__ dst,
                                   const float* __restrict__ eattr,
                                   const float* __restrict__ xl, const float* __restrict__ xr,
                                   const float* __restrict__ We, const float* __restrict__ att,
                                   const float* __restrict__ loop_attr,
                                   const float* __restrict__ deg,
                                   float* __restrict__ logits, unsigned* __restrict__ mxkey,
                                   int E, int N) {
    __shared__ float sWe[HIDV * EDIM];
    __shared__ float sAtt[HIDV];
    for (int t = threadIdx.x; t < HIDV * EDIM; t += blockDim.x) sWe[t] = We[t];
    for (int t = threadIdx.x; t < HIDV; t += blockDim.x) sAtt[t] = att[t];
    __syncthreads();
    int i = blockIdx.x * blockDim.x + threadIdx.x;
    if (i >= E + N) return;
    int s, d;
    float ea[EDIM];
    if (i < E) {
        s = src[i]; d = dst[i];
#pragma unroll
        for (int c = 0; c < EDIM; ++c) ea[c] = eattr[(size_t)i * EDIM + c];
    } else {                       // self loop, fill_value='mean'
        s = d = i - E;
        float dg = deg[d]; dg = dg > 1.0f ? dg : 1.0f;
#pragma unroll
        for (int c = 0; c < EDIM; ++c) ea[c] = loop_attr[(size_t)d * EDIM + c] / dg;
    }
    const float* pl = xl + (size_t)s * HIDV;
    const float* pr = xr + (size_t)d * HIDV;
    float logit = 0.0f;
#pragma unroll 4
    for (int j = 0; j < HIDV; ++j) {
        float mm = pl[j] + pr[j];
#pragma unroll
        for (int c = 0; c < EDIM; ++c) mm += ea[c] * sWe[j * EDIM + c];
        mm = mm > 0.0f ? mm : NEG_SLOPE * mm;
        logit += mm * sAtt[j];
    }
    logits[i] = logit;
    atomicMax(mxkey + d, fkey(logit));
}

// ---------------- kernel 6: softmax denominator -------------------------------
__global__ void edge_denom_kernel(const int* __restrict__ dst, const float* __restrict__ logits,
                                  const unsigned* __restrict__ mxkey,
                                  float* __restrict__ denom, int E, int N) {
    int i = blockIdx.x * blockDim.x + threadIdx.x;
    if (i >= E + N) return;
    int d = (i < E) ? dst[i] : (i - E);
    float e = __expf(logits[i] - funkey(mxkey[d]));
    atomicAdd(denom + d, e);
}

// ---------------- kernel 7: alpha-weighted scatter aggregation ----------------
__global__ void edge_scatter_kernel(const int* __restrict__ src, const int* __restrict__ dst,
                                    const float* __restrict__ logits,
                                    const unsigned* __restrict__ mxkey,
                                    const float* __restrict__ denom,
                                    const float* __restrict__ xl,
                                    float* __restrict__ agg, int E, int N) {
    int i = blockIdx.x * blockDim.x + threadIdx.x;
    if (i >= E + N) return;
    int s, d;
    if (i < E) { s = src[i]; d = dst[i]; } else { s = d = i - E; }
    float alpha = __expf(logits[i] - funkey(mxkey[d])) / denom[d];
    const float* pl = xl + (size_t)s * HIDV;
    float* pa = agg + (size_t)d * HIDV;
#pragma unroll 8
    for (int j = 0; j < HIDV; ++j)
        atomicAdd(pa + j, pl[j] * alpha);
}

// ---------------- kernel 8: h = relu(agg + bias_g); q = h @ W2^T + b2 ---------
__global__ void final_kernel(const float* __restrict__ agg, const float* __restrict__ bias_g,
                             const float* __restrict__ W2, const float* __restrict__ b2,
                             float* __restrict__ q, int N) {
    __shared__ float sW2[NACT * HIDV];
    __shared__ float sb2[NACT];
    __shared__ float sbg[HIDV];
    for (int t = threadIdx.x; t < NACT * HIDV; t += blockDim.x) sW2[t] = W2[t];
    for (int t = threadIdx.x; t < NACT; t += blockDim.x) sb2[t] = b2[t];
    for (int t = threadIdx.x; t < HIDV; t += blockDim.x) sbg[t] = bias_g[t];
    __syncthreads();
    int i = blockIdx.x * blockDim.x + threadIdx.x;
    if (i >= N) return;
    float h[HIDV];
#pragma unroll
    for (int j = 0; j < HIDV; ++j) {
        float v = agg[(size_t)i * HIDV + j] + sbg[j];
        h[j] = v > 0.0f ? v : 0.0f;
    }
#pragma unroll
    for (int o = 0; o < NACT; ++o) {
        float acc = sb2[o];
#pragma unroll
        for (int j = 0; j < HIDV; ++j) acc += h[j] * sW2[o * HIDV + j];
        q[(size_t)i * NACT + o] = acc;
    }
}

// ---------------- launcher ----------------------------------------------------
extern "C" void kernel_launch(void* const* d_in, const int* in_sizes, int n_in,
                              void* d_out, int out_size, void* d_ws, size_t ws_size,
                              hipStream_t stream) {
    const float* inputs = (const float*)d_in[0];
    const float* hidden = (const float*)d_in[1];
    const int*   eidx   = (const int*)  d_in[2];
    const float* eattr  = (const float*)d_in[3];
    const float* W1     = (const float*)d_in[4];
    const float* b1     = (const float*)d_in[5];
    const float* w_ih   = (const float*)d_in[6];
    const float* w_hh   = (const float*)d_in[7];
    const float* b_ih   = (const float*)d_in[8];
    const float* b_hh   = (const float*)d_in[9];
    const float* Wl     = (const float*)d_in[10];
    const float* bl     = (const float*)d_in[11];
    const float* Wr     = (const float*)d_in[12];
    const float* br     = (const float*)d_in[13];
    const float* We     = (const float*)d_in[14];
    const float* att    = (const float*)d_in[15];
    const float* bias_g = (const float*)d_in[16];
    const float* W2     = (const float*)d_in[17];
    const float* b2     = (const float*)d_in[18];

    const int N = in_sizes[1] / H2;          // 100000
    const int E = in_sizes[3] / EDIM;        // 1600000
    const int* src = eidx;
    const int* dst = eidx + E;

    float* qout  = (float*)d_out;            // [N, 14]
    float* hsout = qout + (size_t)N * NACT;  // [N, 128]

    char* ws = (char*)d_ws;
    __bf16*   wbf    = (__bf16*)ws;                              // 196608 B
    float*    xl     = (float*)(ws + WBF_TOT * 2);
    float*    xr     = xl  + (size_t)N * HIDV;
    float*    agg    = xr  + (size_t)N * HIDV;
    float*    logits = agg + (size_t)N * HIDV;
    float*    deg    = logits + (size_t)(E + N);
    float*    loopat = deg + N;
    unsigned* mxkey  = (unsigned*)(loopat + (size_t)N * EDIM);
    float*    denom  = (float*)(mxkey + N);

    const int T = 256;
    const int nTiles = N / 16;

    prep_weights<<<(WBF_TOT + T - 1) / T, T, 0, stream>>>(W1, w_ih, w_hh, Wl, Wr, wbf);
    init_ws<<<((size_t)N * HIDV + T - 1) / T, T, 0, stream>>>(agg, deg, loopat, denom, mxkey, N);

    node_kernel<<<(nTiles + 7) / 8, T, 0, stream>>>(
        inputs, hidden, b1, b_ih, b_hh, bl, br, wbf, hsout, xl, xr, nTiles);

    edge_deg_kernel<<<(E + T - 1) / T, T, 0, stream>>>(dst, eattr, deg, loopat, E);
    edge_logits_kernel<<<(E + N + T - 1) / T, T, 0, stream>>>(
        src, dst, eattr, xl, xr, We, att, loopat, deg, logits, mxkey, E, N);
    edge_denom_kernel<<<(E + N + T - 1) / T, T, 0, stream>>>(dst, logits, mxkey, denom, E, N);
    edge_scatter_kernel<<<(E + N + T - 1) / T, T, 0, stream>>>(
        src, dst, logits, mxkey, denom, xl, agg, E, N);

    final_kernel<<<(N + T - 1) / T, T, 0, stream>>>(agg, bias_g, W2, b2, qout, N);
}
